// TreeModel_6176162972299
// MI455X (gfx1250) — matrix operand order repaired and checked by reading, not compiled, not run on hardware
//
#include <hip/hip_runtime.h>

typedef __attribute__((ext_vector_type(16))) _Float16 v16h;
typedef __attribute__((ext_vector_type(8)))  float    v8f;

#define HH     32     // LSTM hidden size
#define GATES  128    // 4*HH
#define LSEQ   256
#define NR     32     // rows (sequences) per block in kernel 1
#define TCH    64     // timestep chunk staged in LDS
#define BATCH  64
#define MNODES 64

// Branch-free activations: v_exp_f32 + v_rcp_f32, no libm branches / div-fixup.
__device__ __forceinline__ float sigm(float x) {
    return __builtin_amdgcn_rcpf(1.f + __expf(-x));
}
__device__ __forceinline__ float tanh_fast(float x) {
    return 2.f * __builtin_amdgcn_rcpf(1.f + __expf(-2.f * x)) - 1.f;
}

// ---- WMMA fragment loaders -------------------------------------------------
// B fragment (K x 16 columns) from row-major f32 weight matrix w[GATES][Kdim].
// ISA 16-bit B layout: lane = column, lanes 0-15 hold K(local) 0..7 and 16..23,
// lanes 16-31 hold K(local) 8..15 and 24..31 (2 packed f16 per VGPR).
__device__ __forceinline__ v16h load_bfrag(const float* __restrict__ w, int Kdim,
                                           int colBase, int kOff, int lane) {
    int col = colBase + (lane & 15);
    int k0  = kOff + ((lane >> 4) << 3);
    const float* p = w + col * Kdim + k0;
    v16h b;
#pragma unroll
    for (int e = 0; e < 16; ++e) b[e] = (_Float16)p[(e & 7) + ((e >> 3) << 4)];
    return b;
}

// A fragment (16 rows x 32 K) from a row-major f16 LDS matrix.
// ISA 16-bit A layout: lanes 0-15 -> K 0..7 (elems 0-7) and 16..23 (elems 8-15);
// lanes 16-31 -> K 8..15 and 24..31.  Each lane = two contiguous b128 DS loads.
__device__ __forceinline__ v16h load_afrag_lds(const _Float16* s, int rowStride,
                                               int rowBase, int kOff, int lane) {
    int row  = rowBase + (lane & 15);
    int half = lane >> 4;
    const _Float16* p = s + row * rowStride + kOff;
    v16h a;
#pragma unroll
    for (int e = 0; e < 8; ++e) a[e] = p[half * 8 + e];
#pragma unroll
    for (int e = 0; e < 8; ++e) a[8 + e] = p[16 + half * 8 + e];
    return a;
}

// Same, but sourced from a row-major f32 global matrix (converted to f16).
__device__ __forceinline__ v16h load_afrag_g32(const float* __restrict__ g, int rowStride,
                                               int rowBase, int kOff, int lane) {
    int row  = rowBase + (lane & 15);
    int half = lane >> 4;
    const float* p = g + row * rowStride + kOff;
    v16h a;
#pragma unroll
    for (int e = 0; e < 8; ++e) a[e] = (_Float16)p[half * 8 + e];
#pragma unroll
    for (int e = 0; e < 8; ++e) a[8 + e] = (_Float16)p[16 + half * 8 + e];
    return a;
}

// ============================================================================
// Kernel 1: lstm1 forward scan over L=256 (4096 sequences), plus the single
// backward step at t=L-1.  Writes a = [hf|hb] : float[4096][64].
//
// Per step critical path: ds_load A-frag -> 1 WMMA (K=32 == full hidden) ->
// 8 ds_store -> barrier -> straight-line gate math (x-term folded here, since
// hid = tid&31 is thread-invariant) -> barrier.  x staged in LDS per 64-step
// chunk with coalesced loads + global_prefetch of the next chunk.
// ============================================================================
__global__ __launch_bounds__(256)
void lstm1_kernel(const float* __restrict__ x,        // [N=4096][L=256]
                  const float* __restrict__ w_ih1f,   // [128][1]
                  const float* __restrict__ w_hh1f,   // [128][32]
                  const float* __restrict__ b_ih1f, const float* __restrict__ b_hh1f,
                  const float* __restrict__ w_ih1b,   // [128][1]
                  const float* __restrict__ b_ih1b, const float* __restrict__ b_hh1b,
                  float* __restrict__ a_out)          // [4096][64]
{
    __shared__ _Float16 sh_h[NR * HH];
    __shared__ float    sh_c[NR * HH];
    __shared__ float    sh_g[NR * GATES];
    __shared__ float    sh_x[NR * TCH];

    const int tid     = threadIdx.x;
    const int lane    = tid & 31;
    const int wid     = tid >> 5;          // 0..7 : gate-column tile
    const int colBase = wid * 16;
    const int col     = colBase + (lane & 15);
    const int half    = lane >> 4;
    const int rowG0   = blockIdx.x * NR;

    // Recurrent weight B-fragment pinned in VGPRs for all 256 steps (K=32).
    const v16h bhh = load_bfrag(w_hh1f, HH, colBase, 0, lane);

    // Update-phase constants: hid = tid&31 is invariant across the e-loop.
    const int hid_u  = tid & 31;
    const int rowu0  = tid >> 5;           // rows rowu0 + 8*e
    const float wih_i = w_ih1f[hid_u],      wih_f = w_ih1f[32 + hid_u];
    const float wih_g = w_ih1f[64 + hid_u], wih_o = w_ih1f[96 + hid_u];
    const float bs_i = b_ih1f[hid_u]      + b_hh1f[hid_u];
    const float bs_f = b_ih1f[32 + hid_u] + b_hh1f[32 + hid_u];
    const float bs_g = b_ih1f[64 + hid_u] + b_hh1f[64 + hid_u];
    const float bs_o = b_ih1f[96 + hid_u] + b_hh1f[96 + hid_u];

    for (int e = tid; e < NR * HH; e += 256) { sh_h[e] = (_Float16)0.f; sh_c[e] = 0.f; }

    for (int tc = 0; tc < LSEQ; tc += TCH) {
        __syncthreads();   // prior update phase / init done before restaging x
        // Stage x[rowG0..+NR)[tc..tc+TCH) : coalesced, 8 loads per thread.
#pragma unroll
        for (int e = 0; e < (NR * TCH) / 256; ++e) {
            int i   = e * 256 + tid;
            int row = i >> 6, tt = i & 63;
            const float* gp = &x[(rowG0 + row) * LSEQ + tc + tt];
            sh_x[row * TCH + tt] = *gp;
            if (tc + TCH < LSEQ) __builtin_prefetch(gp + TCH, 0, 0);
        }
        __syncthreads();

        for (int t = 0; t < TCH; ++t) {
            // ---- matmul phase: g_tile = h @ Whh^T (one WMMA per row tile) ----
#pragma unroll
            for (int rt = 0; rt < NR / 16; ++rt) {
                v16h a   = load_afrag_lds(sh_h, HH, rt * 16, 0, lane);
                v8f  acc = {};
                acc = __builtin_amdgcn_wmma_f32_16x16x32_f16(false, a, false, bhh,
                                                             (short)0, acc, false, false);
#pragma unroll
                for (int r = 0; r < 8; ++r)
                    sh_g[(rt * 16 + half * 8 + r) * GATES + col] = acc[r];
            }
            __syncthreads();
            // ---- elementwise gate phase (x-term + bias folded in here) ------
#pragma unroll
            for (int e = 0; e < (NR * HH) / 256; ++e) {
                int row = e * 8 + rowu0;
                int idx = row * HH + hid_u;
                float xv = sh_x[row * TCH + t];
                const float* gr = sh_g + row * GATES + hid_u;
                float gi = sigm     (gr[0]  + xv * wih_i + bs_i);
                float gf = sigm     (gr[32] + xv * wih_f + bs_f);
                float gg = tanh_fast(gr[64] + xv * wih_g + bs_g);
                float go = sigm     (gr[96] + xv * wih_o + bs_o);
                float c  = gf * sh_c[idx] + gi * gg;
                sh_c[idx] = c;
                sh_h[idx] = (_Float16)(go * tanh_fast(c));
            }
            if (t != TCH - 1) __syncthreads();
        }
    }
    __syncthreads();

    // emit hf, and the one-step backward hidden (zero initial state: f-gate dead)
#pragma unroll
    for (int e = 0; e < (NR * HH) / 256; ++e) {
        int row = e * 8 + rowu0;
        int idx = row * HH + hid_u;
        int n   = rowG0 + row;
        a_out[n * 64 + hid_u] = (float)sh_h[idx];
        float xl = sh_x[row * TCH + (TCH - 1)];   // last chunk holds t = L-1
        float gi = sigm     (xl * w_ih1b[hid_u]      + b_ih1b[hid_u]      + b_hh1b[hid_u]);
        float gg = tanh_fast(xl * w_ih1b[64 + hid_u] + b_ih1b[64 + hid_u] + b_hh1b[64 + hid_u]);
        float go = sigm     (xl * w_ih1b[96 + hid_u] + b_ih1b[96 + hid_u] + b_hh1b[96 + hid_u]);
        a_out[n * 64 + 32 + hid_u] = go * tanh_fast(gi * gg);
    }
}

// ============================================================================
// Kernel 2: sequential tree chain (63 node steps) + final linear + softmax.
// Single workgroup of 8 waves; weight fragments resident in VGPRs.
// ============================================================================
__global__ __launch_bounds__(256)
void lstm2_kernel(const float* __restrict__ a_ws,     // [B=64][M=64][64]
                  const float* __restrict__ w_ih2f,   // [128][64]
                  const float* __restrict__ w_hh2f,   // [128][32]
                  const float* __restrict__ b_ih2f, const float* __restrict__ b_hh2f,
                  const float* __restrict__ w_ih2b,   // [128][64]
                  const float* __restrict__ b_ih2b, const float* __restrict__ b_hh2b,
                  const float* __restrict__ w_lin,    // [2][64]
                  const float* __restrict__ b_lin,    // [2]
                  float* __restrict__ out)            // [64][2]
{
    __shared__ _Float16 sh_f [BATCH * 64];   // current node feature f_prev
    __shared__ _Float16 sh_nf[BATCH * 64];   // next node feature (built this iter)
    __shared__ _Float16 sh_h [BATCH * HH];   // fwd step-1 hidden
    __shared__ float    sh_c [BATCH * HH];   // fwd step-1 cell
    __shared__ float    sh_g [BATCH * GATES];

    const int tid     = threadIdx.x;
    const int lane    = tid & 31;
    const int wid     = tid >> 5;
    const int colBase = wid * 16;
    const int col     = colBase + (lane & 15);
    const int half    = lane >> 4;

    // Weight fragments (held in registers for all 63 iterations).
    const v16h Bf0 = load_bfrag(w_ih2f, 64, colBase, 0,  lane);
    const v16h Bf1 = load_bfrag(w_ih2f, 64, colBase, 32, lane);
    const v16h Bfh = load_bfrag(w_hh2f, HH, colBase, 0,  lane);
    const v16h Bb0 = load_bfrag(w_ih2b, 64, colBase, 0,  lane);
    const v16h Bb1 = load_bfrag(w_ih2b, 64, colBase, 32, lane);
    const float biasF = b_ih2f[col] + b_hh2f[col];
    const float biasB = b_ih2b[col] + b_hh2b[col];

    // g = A @ W^T from a global node slice a[:, j] (K=64), zero hidden state
    auto gemm_node = [&](int j, const v16h& B0, const v16h& B1, float bias) {
#pragma unroll
        for (int rt = 0; rt < 4; ++rt) {
            v16h A0 = load_afrag_g32(a_ws + j * 64, MNODES * 64, rt * 16, 0,  lane);
            v16h A1 = load_afrag_g32(a_ws + j * 64, MNODES * 64, rt * 16, 32, lane);
            v8f acc = {};
            acc = __builtin_amdgcn_wmma_f32_16x16x32_f16(false, A0, false, B0, (short)0, acc, false, false);
            acc = __builtin_amdgcn_wmma_f32_16x16x32_f16(false, A1, false, B1, (short)0, acc, false, false);
#pragma unroll
            for (int r = 0; r < 8; ++r)
                sh_g[(rt * 16 + half * 8 + r) * GATES + col] = acc[r] + bias;
        }
        __syncthreads();
    };

    // g = f_prev @ Wih^T (+ h @ Whh^T) ; f_prev in LDS (K=64)
    auto gemm_f = [&](const v16h& B0, const v16h& B1, float bias, bool withH) {
#pragma unroll
        for (int rt = 0; rt < 4; ++rt) {
            v16h A0 = load_afrag_lds(sh_f, 64, rt * 16, 0,  lane);
            v16h A1 = load_afrag_lds(sh_f, 64, rt * 16, 32, lane);
            v8f acc = {};
            acc = __builtin_amdgcn_wmma_f32_16x16x32_f16(false, A0, false, B0, (short)0, acc, false, false);
            acc = __builtin_amdgcn_wmma_f32_16x16x32_f16(false, A1, false, B1, (short)0, acc, false, false);
            if (withH) {
                v16h Ah = load_afrag_lds(sh_h, HH, rt * 16, 0, lane);
                acc = __builtin_amdgcn_wmma_f32_16x16x32_f16(false, Ah, false, Bfh, (short)0, acc, false, false);
            }
#pragma unroll
            for (int r = 0; r < 8; ++r)
                sh_g[(rt * 16 + half * 8 + r) * GATES + col] = acc[r] + bias;
        }
        __syncthreads();
    };

    // gates -> (c, h) elementwise; h written as f16 with row stride / column offset
    auto update = [&](const float* c_in, float* c_out, _Float16* h_out, int hstride, int hoff) {
#pragma unroll
        for (int e = 0; e < (BATCH * HH) / 256; ++e) {
            int idx = e * 256 + tid;
            int row = idx >> 5, hid = idx & 31;
            const float* gr = sh_g + row * GATES + hid;
            float gi = sigm(gr[0]);
            float gf = sigm(gr[32]);
            float gg = tanh_fast(gr[64]);
            float go = sigm(gr[96]);
            float c  = gi * gg + (c_in ? gf * c_in[idx] : 0.f);
            if (c_out) c_out[idx] = c;
            h_out[row * hstride + hoff + hid] = (_Float16)(go * tanh_fast(c));
        }
        __syncthreads();
    };

    // ---- leaf node (j = M-1): one fwd + one bwd step from zero state --------
    gemm_node(MNODES - 1, Bf0, Bf1, biasF);
    update(nullptr, nullptr, sh_f, 64, 0);
    gemm_node(MNODES - 1, Bb0, Bb1, biasB);
    update(nullptr, nullptr, sh_f, 64, 32);

    // ---- internal nodes j = M-2 .. 0 ----------------------------------------
    for (int j = MNODES - 2; j >= 0; --j) {
        // backward: hb2 = step(0,0,f_prev)   (Whh2b unused: zero hidden state)
        gemm_f(Bb0, Bb1, biasB, false);
        update(nullptr, nullptr, sh_nf, 64, 32);
        // forward step 1: (h,c) = step(0,0, a[:,j])
        gemm_node(j, Bf0, Bf1, biasF);
        update(nullptr, sh_c, sh_h, HH, 0);
        // forward step 2: h2 = step(h,c, f_prev)
        gemm_f(Bf0, Bf1, biasF, true);
        update(sh_c, nullptr, sh_nf, 64, 0);
        // f_prev <- new f
        for (int e = tid; e < BATCH * 64; e += 256) sh_f[e] = sh_nf[e];
        __syncthreads();
    }

    // ---- final linear + softmax --------------------------------------------
    if (tid < BATCH) {
        float z0 = b_lin[0], z1 = b_lin[1];
#pragma unroll
        for (int k = 0; k < 64; ++k) {
            float fv = (float)sh_f[tid * 64 + k];
            z0 += fv * w_lin[k];
            z1 += fv * w_lin[64 + k];
        }
        float m  = fmaxf(z0, z1);
        float e0 = __expf(z0 - m), e1 = __expf(z1 - m);
        float s  = e0 + e1;
        out[tid * 2 + 0] = e0 * __builtin_amdgcn_rcpf(s);
        out[tid * 2 + 1] = e1 * __builtin_amdgcn_rcpf(s);
    }
}

// ============================================================================
extern "C" void kernel_launch(void* const* d_in, const int* in_sizes, int n_in,
                              void* d_out, int out_size, void* d_ws, size_t ws_size,
                              hipStream_t stream) {
    const float* x      = (const float*)d_in[0];
    const float* w_ih1f = (const float*)d_in[1];
    const float* w_hh1f = (const float*)d_in[2];
    const float* b_ih1f = (const float*)d_in[3];
    const float* b_hh1f = (const float*)d_in[4];
    const float* w_ih1b = (const float*)d_in[5];
    /* d_in[6] = w_hh1b : unused (backward lstm1 steps from zero state) */
    const float* b_ih1b = (const float*)d_in[7];
    const float* b_hh1b = (const float*)d_in[8];
    const float* w_ih2f = (const float*)d_in[9];
    const float* w_hh2f = (const float*)d_in[10];
    const float* b_ih2f = (const float*)d_in[11];
    const float* b_hh2f = (const float*)d_in[12];
    const float* w_ih2b = (const float*)d_in[13];
    /* d_in[14] = w_hh2b : unused (backward lstm2 steps from zero state) */
    const float* b_ih2b = (const float*)d_in[15];
    const float* b_hh2b = (const float*)d_in[16];
    const float* w_lin  = (const float*)d_in[17];
    const float* b_lin  = (const float*)d_in[18];

    float* a_ws = (float*)d_ws;   // [4096][64] node features (1 MB)

    lstm1_kernel<<<(64 * MNODES) / NR, 256, 0, stream>>>(
        x, w_ih1f, w_hh1f, b_ih1f, b_hh1f, w_ih1b, b_ih1b, b_hh1b, a_ws);

    lstm2_kernel<<<1, 256, 0, stream>>>(
        a_ws, w_ih2f, w_hh2f, b_ih2f, b_hh2f, w_ih2b, b_ih2b, b_hh2b,
        w_lin, b_lin, (float*)d_out);
}
